// FeatureGNN_8830452761019
// MI455X (gfx1250) — matrix operand order repaired and checked by reading, not compiled, not run on hardware
//
#include <hip/hip_runtime.h>
#include <hip/hip_bf16.h>

typedef __attribute__((ext_vector_type(16))) __bf16 v16bf;
typedef __attribute__((ext_vector_type(8)))  __bf16 v8bf;
typedef __attribute__((ext_vector_type(8)))  float  v8f;

static constexpr int N_NODES = 131072;
static constexpr int N_EDGES = 2097152;
static constexpr int N_GRAPH = 1024;

__device__ __forceinline__ unsigned short f2bf_rne(float f) {
  unsigned u = __float_as_uint(f);
  u += 0x7FFFu + ((u >> 16) & 1u);      // round-to-nearest-even to bf16
  return (unsigned short)(u >> 16);
}

// ---------------- graph normalization ----------------
__global__ void k_fill(float* __restrict__ p, float v, int n) {
  int i = blockIdx.x * blockDim.x + threadIdx.x;
  if (i < n) p[i] = v;
}

__global__ void k_scatter_deg(const int* __restrict__ ei, const float* __restrict__ w,
                              float* __restrict__ deg) {
  int e = blockIdx.x * blockDim.x + threadIdx.x;
  if (e < N_EDGES) atomicAdd(&deg[ei[N_EDGES + e]], w[e]);   // col = edge_index[1]
}

__global__ void k_dis(float* __restrict__ d, int n) {
  int i = blockIdx.x * blockDim.x + threadIdx.x;
  if (i < n) { float v = d[i]; d[i] = (v > 0.f) ? rsqrtf(v) : 0.f; }
}

__global__ void k_norm(const int* __restrict__ ei, const float* __restrict__ w,
                       const float* __restrict__ dis, float* __restrict__ norm) {
  int e = blockIdx.x * blockDim.x + threadIdx.x;
  if (e < N_EDGES) norm[e] = dis[ei[e]] * w[e] * dis[ei[N_EDGES + e]];
}

// ---- layer 1 transform (K=1) fused with aggregation self-loop init ----
__global__ void k_l1_fused(const float* __restrict__ x, const float* __restrict__ W1,
                           const float* __restrict__ dis, const float* __restrict__ b1,
                           float* __restrict__ t, float* __restrict__ agg) {
  long long gid = (long long)blockIdx.x * blockDim.x + threadIdx.x;  // N*128
  int i = (int)(gid >> 7), j = (int)(gid & 127);
  float tv = x[i] * W1[j];
  float dv = dis[i];
  t[gid] = tv;
  agg[gid] = dv * dv * tv + b1[j];          // self-loop norm = dis[i]^2, + bias
}

// ---------------- weight packing into WMMA B-fragment order ----------------
// P[((tileN*ksteps + s)*32 + lane)*16 + i] = W[k*DO + n] (bf16)
// with k = s*32 + (lane>>4)*16 + i,  n = tileN*16 + (lane&15)
__global__ void k_pack_w(const float* __restrict__ W, unsigned short* __restrict__ P,
                         int K, int DO) {
  int idx = blockIdx.x * blockDim.x + threadIdx.x;
  if (idx >= K * DO) return;
  int i    = idx & 15;
  int lane = (idx >> 4) & 31;
  int rest = idx >> 9;
  int ksteps = K >> 5;
  int s = rest % ksteps;
  int tn = rest / ksteps;
  int k = s * 32 + ((lane >> 4) << 4) + i;
  int n = (tn << 4) + (lane & 15);
  P[idx] = f2bf_rne(W[(size_t)k * DO + n]);
}

// ---- bf16 WMMA GEMM fused with aggregation self-loop init:
//      C[nrows x DO] = A @ W ;  AG = dis[row]^2 * C + bias[col]
__global__ __launch_bounds__(256) void k_gemm_wmma_fused(
    const unsigned short* __restrict__ A, const unsigned short* __restrict__ BP,
    float* __restrict__ C, float* __restrict__ AG,
    const float* __restrict__ dis, const float* __restrict__ bias,
    int nrows, int K, int DO) {
  const int lane  = threadIdx.x & 31;
  const int wave  = blockIdx.x * (blockDim.x >> 5) + (threadIdx.x >> 5);
  const int tilesN = DO >> 4;
  const int tileM = wave / tilesN;
  const int tileN = wave - tileM * tilesN;
  if (tileM * 16 >= nrows) return;
  const int ksteps = K >> 5;
  const int row = tileM * 16 + (lane & 15);
  const int kb1 = (lane & 16) ? 16 : 0;         // contiguous A layout: k = 16h + i (matches packed B)
  const unsigned short* arow = A + (size_t)row * K;
  const unsigned short* bp   = BP + ((size_t)tileN * ksteps) * 512 + (size_t)lane * 16;
  v8f acc = {};
  for (int s = 0; s < ksteps; ++s) {
    v8bf alo = *(const v8bf*)(const void*)(arow + s * 32 + kb1);
    v8bf ahi = *(const v8bf*)(const void*)(arow + s * 32 + kb1 + 8);
    v16bf a = __builtin_shufflevector(alo, ahi, 0,1,2,3,4,5,6,7,8,9,10,11,12,13,14,15);
    v16bf b = *(const v16bf*)(const void*)(bp + (size_t)s * 512);
    acc = __builtin_amdgcn_wmma_f32_16x16x32_bf16(false, a, false, b,
                                                  (short)0, acc, false, false);
  }
  const int mbase = tileM * 16 + ((lane >> 4) << 3);
  const int col   = tileN * 16 + (lane & 15);
  const float bcol = bias[col];
  float* crow = C  + (size_t)mbase * DO + col;
  float* grow = AG + (size_t)mbase * DO + col;
  #pragma unroll
  for (int v = 0; v < 8; ++v) {
    float val = acc[v];
    float dv  = dis[mbase + v];
    crow[(size_t)v * DO] = val;
    grow[(size_t)v * DO] = dv * dv * val + bcol;
  }
}

// ---------------- edge scatter (atomic accumulate), D = 4 << qshift ----------------
__global__ void k_agg_edges(const int* __restrict__ ei, const float* __restrict__ norm,
                            const float* __restrict__ t, float* __restrict__ out,
                            int qshift) {
  long long gid = (long long)blockIdx.x * blockDim.x + threadIdx.x;
  long long total = ((long long)N_EDGES) << qshift;
  if (gid >= total) return;
  int e = (int)(gid >> qshift);
  int q = (int)(gid & ((1 << qshift) - 1));
  float nv = norm[e];
  int r = ei[e];
  int c = ei[N_EDGES + e];
  int dshift = qshift + 2;
  const float4 tv = ((const float4*)(t + ((size_t)r << dshift)))[q];
  float* op = out + ((size_t)c << dshift) + ((size_t)q << 2);
  atomicAdd(op + 0, nv * tv.x);
  atomicAdd(op + 1, nv * tv.y);
  atomicAdd(op + 2, nv * tv.z);
  atomicAdd(op + 3, nv * tv.w);
}

// ---- fused ReLU + f32->bf16 convert (activations consumed only by next GEMM) ----
__global__ void k_relu_f2bf(const float* __restrict__ in, unsigned short* __restrict__ out,
                            long long n) {
  long long i = (long long)blockIdx.x * blockDim.x + threadIdx.x;
  if (i < n) out[i] = f2bf_rne(fmaxf(in[i], 0.f));
}

// ---------------- pooling (ReLU of layer 3 fused here) + MLP head ----------------
__global__ void k_pool_scatter(const int* __restrict__ batch, const float* __restrict__ h,
                               float* __restrict__ pooled, float* __restrict__ cnt) {
  long long gid = (long long)blockIdx.x * blockDim.x + threadIdx.x;  // N*128
  int i = (int)(gid >> 7), j = (int)(gid & 127);
  int b = batch[i];
  atomicAdd(&pooled[(size_t)b * 128 + j], fmaxf(h[gid], 0.f));
  if (j == 0) atomicAdd(&cnt[b], 1.0f);
}

__global__ void k_pool_fin(float* __restrict__ pooled, const float* __restrict__ cnt) {
  int gid = blockIdx.x * blockDim.x + threadIdx.x;   // G*128
  int g = gid >> 7;
  pooled[gid] /= fmaxf(cnt[g], 1.0f);
}

__global__ void k_mlp1(const float* __restrict__ pooled, const float* __restrict__ Wf1,
                       const float* __restrict__ bf1, float* __restrict__ z) {
  int gid = blockIdx.x * blockDim.x + threadIdx.x;   // G*32
  int g = gid >> 5, k = gid & 31;
  float s = bf1[k];
  const float* p = pooled + (size_t)g * 128;
  #pragma unroll 4
  for (int i = 0; i < 128; ++i) s += p[i] * Wf1[i * 32 + k];
  z[gid] = fmaxf(s, 0.f);
}

__global__ void k_mlp2(const float* __restrict__ z, const float* __restrict__ Wf2,
                       const float* __restrict__ bf2, float* __restrict__ out) {
  int gid = blockIdx.x * blockDim.x + threadIdx.x;   // G*2
  if (gid >= N_GRAPH * 2) return;
  int g = gid >> 1, o = gid & 1;
  float s = bf2[o];
  #pragma unroll
  for (int i = 0; i < 32; ++i) s += z[g * 32 + i] * Wf2[i * 2 + o];
  out[gid] = s;
}

// ---------------- workspace layout ----------------
static constexpr size_t aln(size_t x) { return (x + 255) & ~(size_t)255; }
static constexpr size_t OFF_DIS  = 0;
static constexpr size_t OFF_NORM = OFF_DIS  + aln((size_t)N_NODES * 4);
static constexpr size_t OFF_HA   = OFF_NORM + aln((size_t)N_EDGES * 4);
static constexpr size_t OFF_HB   = OFF_HA   + aln((size_t)N_NODES * 256 * 4);
static constexpr size_t OFF_HBF  = OFF_HB   + aln((size_t)N_NODES * 256 * 4);
static constexpr size_t OFF_WP2  = OFF_HBF  + aln((size_t)N_NODES * 256 * 2);
static constexpr size_t OFF_WP3  = OFF_WP2  + aln((size_t)128 * 256 * 2);
static constexpr size_t OFF_POOL = OFF_WP3  + aln((size_t)256 * 128 * 2);
static constexpr size_t OFF_CNT  = OFF_POOL + aln((size_t)N_GRAPH * 128 * 4);
static constexpr size_t OFF_Z    = OFF_CNT  + aln((size_t)N_GRAPH * 4);

extern "C" void kernel_launch(void* const* d_in, const int* in_sizes, int n_in,
                              void* d_out, int out_size, void* d_ws, size_t ws_size,
                              hipStream_t stream) {
  const float* x    = (const float*)d_in[0];
  const int*   ei   = (const int*)d_in[1];     // [2, E]
  const float* ew   = (const float*)d_in[2];
  const int*   bat  = (const int*)d_in[3];
  const float* W1   = (const float*)d_in[4];
  const float* b1   = (const float*)d_in[5];
  const float* W2   = (const float*)d_in[6];
  const float* b2   = (const float*)d_in[7];
  const float* W3   = (const float*)d_in[8];
  const float* b3   = (const float*)d_in[9];
  const float* Wf1  = (const float*)d_in[10];
  const float* bf1  = (const float*)d_in[11];
  const float* Wf2  = (const float*)d_in[12];
  const float* bf2  = (const float*)d_in[13];
  float* out = (float*)d_out;

  char* ws = (char*)d_ws;
  float* dis   = (float*)(ws + OFF_DIS);
  float* norm  = (float*)(ws + OFF_NORM);
  float* hA    = (float*)(ws + OFF_HA);
  float* hB    = (float*)(ws + OFF_HB);
  unsigned short* hbf = (unsigned short*)(ws + OFF_HBF);
  unsigned short* wp2 = (unsigned short*)(ws + OFF_WP2);
  unsigned short* wp3 = (unsigned short*)(ws + OFF_WP3);
  float* pooled = (float*)(ws + OFF_POOL);
  float* cnt    = (float*)(ws + OFF_CNT);
  float* z      = (float*)(ws + OFF_Z);

  const int B = 256;
  // --- gcn_norm ---
  k_fill<<<N_NODES / B, B, 0, stream>>>(dis, 1.0f, N_NODES);        // self-loop weight
  k_scatter_deg<<<N_EDGES / B, B, 0, stream>>>(ei, ew, dis);
  k_dis<<<N_NODES / B, B, 0, stream>>>(dis, N_NODES);
  k_norm<<<N_EDGES / B, B, 0, stream>>>(ei, ew, dis, norm);

  // --- weight packs (small, per launch) ---
  k_pack_w<<<(128 * 256) / B, B, 0, stream>>>(W2, wp2, 128, 256);
  k_pack_w<<<(256 * 128) / B, B, 0, stream>>>(W3, wp3, 256, 128);

  const long long n128 = (long long)N_NODES * 128;
  const long long n256 = (long long)N_NODES * 256;

  // --- layer 1: t = x@W1 -> hB, agg-init -> hA (fused); edge scatter; relu+bf16 ---
  k_l1_fused<<<(int)(n128 / B), B, 0, stream>>>(x, W1, dis, b1, hB, hA);
  k_agg_edges<<<(int)(((long long)N_EDGES << 5) / B), B, 0, stream>>>(ei, norm, hB, hA, 5);
  k_relu_f2bf<<<(int)(n128 / B), B, 0, stream>>>(hA, hbf, n128);

  // --- layer 2: WMMA gemm (K=128, DO=256) -> hB + agg-init -> hA; scatter; relu+bf16 ---
  k_gemm_wmma_fused<<<(N_NODES / 16) * (256 / 16) / 8, B, 0, stream>>>(
      hbf, wp2, hB, hA, dis, b2, N_NODES, 128, 256);
  k_agg_edges<<<(int)(((long long)N_EDGES << 6) / B), B, 0, stream>>>(ei, norm, hB, hA, 6);
  k_relu_f2bf<<<(int)(n256 / B), B, 0, stream>>>(hA, hbf, n256);

  // --- layer 3: WMMA gemm (K=256, DO=128) -> hB + agg-init -> hA; scatter ---
  k_gemm_wmma_fused<<<(N_NODES / 16) * (128 / 16) / 8, B, 0, stream>>>(
      hbf, wp3, hB, hA, dis, b3, N_NODES, 256, 128);
  k_agg_edges<<<(int)(((long long)N_EDGES << 5) / B), B, 0, stream>>>(ei, norm, hB, hA, 5);

  // --- global mean pool (layer-3 ReLU fused into the scatter) ---
  k_fill<<<(N_GRAPH * 128) / B, B, 0, stream>>>(pooled, 0.f, N_GRAPH * 128);
  k_fill<<<(N_GRAPH + B - 1) / B, B, 0, stream>>>(cnt, 0.f, N_GRAPH);
  k_pool_scatter<<<(int)(n128 / B), B, 0, stream>>>(bat, hA, pooled, cnt);
  k_pool_fin<<<(N_GRAPH * 128) / B, B, 0, stream>>>(pooled, cnt);

  // --- MLP head ---
  k_mlp1<<<(N_GRAPH * 32) / B, B, 0, stream>>>(pooled, Wf1, bf1, z);
  k_mlp2<<<(N_GRAPH * 2 + B - 1) / B, B, 0, stream>>>(z, Wf2, bf2, out);
}
